// GNN_5781025980771
// MI455X (gfx1250) — matrix operand order repaired and checked by reading, not compiled, not run on hardware
//
#include <hip/hip_runtime.h>
#include <hip/hip_bf16.h>

// ---------------------------------------------------------------------------
// 2-layer GCN for MI455X (gfx1250).
//  - GEMMs via v_wmma_f32_16x16x32_bf16 (wave32, 8 waves/block, LDS-staged A,
//    pre-packed B fragments).
//  - Edge aggregation via global_atomic_add_f32; 51.2MB feature matrices fit
//    in the 192MB L2 so random per-edge traffic stays on-chip.
// ---------------------------------------------------------------------------

#define NNODES 100000
#define NEDGES 1600000
#define DIM    128

typedef __attribute__((ext_vector_type(16))) __bf16 v16bf;
typedef __attribute__((ext_vector_type(8)))  float  v8f;

union FragAB { v16bf v; unsigned int u[8]; uint4 q[2]; };
union FragC  { v8f   v; float f[8]; };

static __device__ __forceinline__ unsigned int pack_bf16(float lo, float hi) {
    unsigned int a = __float_as_uint(lo);
    unsigned int b = __float_as_uint(hi);
    a = (a + 0x7FFFu + ((a >> 16) & 1u)) >> 16;   // round-to-nearest-even
    b = (b + 0x7FFFu + ((b >> 16) & 1u)) >> 16;
    return (a & 0xFFFFu) | (b << 16);
}

// ---------------------------------------------------------------------------
// Utility kernels
// ---------------------------------------------------------------------------
__global__ void zero_f32(float* __restrict__ p, int n) {
    int t = blockIdx.x * blockDim.x + threadIdx.x;
    if (t < n) p[t] = 0.0f;
}

__global__ void deg_count(const long long* __restrict__ col,
                          float* __restrict__ deg, int nedges) {
    int e = blockIdx.x * blockDim.x + threadIdx.x;
    if (e < nedges) atomicAdd(&deg[(int)col[e]], 1.0f);
}

__global__ void make_dinv(float* __restrict__ deg, int n) {
    int t = blockIdx.x * blockDim.x + threadIdx.x;
    if (t < n) deg[t] = __frsqrt_rn(deg[t] + 1.0f);   // in-place: deg -> dinv
}

// Pack W[128x128] fp32 into WMMA B fragments (bf16), layout:
// PW[kt(4)][nt(8)][lane(32)][vgpr(8)] dwords; contiguous K order (k = 16*half + s), same as A:
// VGPR v, lanes 0-15: K = {kt*32 + 2v, +1}, N = lane; lanes 16-31: K += 16.
__global__ void pack_W(const float* __restrict__ W, unsigned int* __restrict__ PW) {
    int idx = blockIdx.x * blockDim.x + threadIdx.x;  // 0..1023
    int kt   = idx >> 8;
    int nt   = (idx >> 5) & 7;
    int lane = idx & 31;
    int N    = lane & 15;
    int half = lane >> 4;
    unsigned int* dst = PW + (size_t)idx * 8;
#pragma unroll
    for (int v = 0; v < 8; ++v) {
        int K = kt * 32 + (half ? 16 : 0) + 2 * v;
        float lo = W[(size_t)K       * DIM + nt * 16 + N];
        float hi = W[(size_t)(K + 1) * DIM + nt * 16 + N];
        dst[v] = pack_bf16(lo, hi);
    }
}

// ---------------------------------------------------------------------------
// WMMA GEMM: C[nrows x 128] = A[nrows x 128] (fp32, converted to bf16) @ W
// One block = 16 rows; wave w computes the 16x16 tile for cols [16w,16w+16).
// ---------------------------------------------------------------------------
__global__ __launch_bounds__(256)
void gemm_wmma_bf16(const float* __restrict__ A,
                    const unsigned int* __restrict__ PW,
                    float* __restrict__ C, int nrows) {
    __shared__ __align__(16) unsigned int lds[16 * 64];  // 16x128 bf16 tile

    const int t    = threadIdx.x;
    const int row0 = blockIdx.x * 16;

    // Stage A tile -> LDS as bf16 pairs (each thread converts 8 floats).
    {
        int e0 = t * 8;                  // element index in 16x128 tile
        int r  = e0 >> 7;
        int c  = e0 & 127;
        const float4* src = (const float4*)(A + (size_t)(row0 + r) * DIM + c);
        float4 a = src[0];
        float4 b = src[1];
        uint4 st;
        st.x = pack_bf16(a.x, a.y);
        st.y = pack_bf16(a.z, a.w);
        st.z = pack_bf16(b.x, b.y);
        st.w = pack_bf16(b.z, b.w);
        ((uint4*)lds)[t] = st;           // dword index t*4 == e0/2
    }
    __syncthreads();

    const int w    = t >> 5;             // wave id = output column tile
    const int lane = t & 31;
    const int M    = lane & 15;
    const int half = lane >> 4;

    FragC acc;
#pragma unroll
    for (int i = 0; i < 8; ++i) acc.f[i] = 0.0f;

    const uint4* lq = (const uint4*)lds;
#pragma unroll
    for (int kt = 0; kt < 4; ++kt) {
        // A fragment (contiguous K order k = 16*half + s, matching packed B): two contiguous b128 LDS reads
        FragAB afr;
        int base = M * 64 + kt * 16;
        afr.q[0] = lq[(base + (half ? 8 : 0))  >> 2];
        afr.q[1] = lq[(base + (half ? 12 : 4)) >> 2];
        // B fragment: pre-packed, contiguous 32B per lane (coalesced b128 x2)
        FragAB bfr;
        const uint4* pw = (const uint4*)(PW + ((size_t)(kt * 8 + w) * 32 + lane) * 8);
        bfr.q[0] = pw[0];
        bfr.q[1] = pw[1];

        acc.v = __builtin_amdgcn_wmma_f32_16x16x32_bf16(
            /*neg_a=*/false, afr.v, /*neg_b=*/false, bfr.v,
            /*c_mod=*/(short)0, acc.v, /*reuse_a=*/false, /*reuse_b=*/false);
    }

    // Epilogue: C/D layout — VGPR v holds row M=v+8*half, col N=lane&15
    float* Cb = C + (size_t)row0 * DIM + w * 16 + M;
    int mrow = half * 8;
#pragma unroll
    for (int v = 0; v < 8; ++v)
        Cb[(size_t)(mrow + v) * DIM] = acc.f[v];
}

// ---------------------------------------------------------------------------
// out[i,:] = xt[i,:] * dinv[i]^2 + bias   (self-loop term; also inits out)
// ---------------------------------------------------------------------------
__global__ __launch_bounds__(256)
void selfloop_bias(const float* __restrict__ xt, const float* __restrict__ dinv,
                   const float* __restrict__ bias, float* __restrict__ out, int n) {
    int t = blockIdx.x * blockDim.x + threadIdx.x;   // over n*32 float4s
    if (t >= n * 32) return;
    int i  = t >> 5;
    int c4 = t & 31;
    float d = dinv[i];
    float s = d * d;
    float4 v = ((const float4*)xt)[t];
    float4 b = ((const float4*)bias)[c4];
    float4 o;
    o.x = v.x * s + b.x;
    o.y = v.y * s + b.y;
    o.z = v.z * s + b.z;
    o.w = v.w * s + b.w;
    ((float4*)out)[t] = o;
}

// ---------------------------------------------------------------------------
// Edge scatter: out[col] += xt[row] * (dinv[row]*dinv[col]).
// One wave per edge, each lane handles a float4 (32*4 = 128 features).
// Atomic f32 adds resolve at L2 (dest matrix fits in 192MB L2).
// ---------------------------------------------------------------------------
__global__ __launch_bounds__(256)
void scatter_edges(const long long* __restrict__ row,
                   const long long* __restrict__ col,
                   const float* __restrict__ dinv,
                   const float* __restrict__ xt,
                   float* __restrict__ out, int nedges) {
    int t = blockIdx.x * blockDim.x + threadIdx.x;
    int e = t >> 5;
    if (e >= nedges) return;
    int lane = t & 31;
    int r = (int)row[e];
    int c = (int)col[e];
    float norm = dinv[r] * dinv[c];
    float4 v = ((const float4*)(xt + (size_t)r * DIM))[lane];
    float* dst = out + (size_t)c * DIM + lane * 4;
    atomicAdd(dst + 0, v.x * norm);
    atomicAdd(dst + 1, v.y * norm);
    atomicAdd(dst + 2, v.z * norm);
    atomicAdd(dst + 3, v.w * norm);
}

// ---------------------------------------------------------------------------
// kernel_launch
// ---------------------------------------------------------------------------
extern "C" void kernel_launch(void* const* d_in, const int* in_sizes, int n_in,
                              void* d_out, int out_size, void* d_ws, size_t ws_size,
                              hipStream_t stream) {
    const float*     x   = (const float*)d_in[0];                 // [100000,128]
    const long long* ei  = (const long long*)d_in[1];             // [2,1600000] int64
    const float*     W1  = (const float*)d_in[2];                 // [128,128]
    const float*     b1  = (const float*)d_in[3];                 // [128]
    const float*     W2  = (const float*)d_in[4];                 // [128,128]
    const float*     b2  = (const float*)d_in[5];                 // [128]
    float*           out = (float*)d_out;                         // [100000,128]

    const long long* row = ei;
    const long long* col = ei + NEDGES;

    // Workspace layout (byte offsets)
    const size_t OFF_PW1 = 524288;      // 512KB
    const size_t OFF_PW2 = 524288 + 65536;
    const size_t OFF_XT  = 1048576;     // 1MB   (51.2MB buffer)
    const size_t OFF_H   = 67108864;    // 64MB  (51.2MB buffer)

    char* ws = (char*)d_ws;
    float*        dinv = (float*)ws;                    // 100000 f32 (deg->dinv)
    unsigned int* PW1  = (unsigned int*)(ws + OFF_PW1); // 32KB packed W1
    unsigned int* PW2  = (unsigned int*)(ws + OFF_PW2); // 32KB packed W2
    float*        XT   = (float*)(ws + OFF_XT);
    float*        H    = (float*)(ws + OFF_H);

    const int TPB = 256;

    // 1) degree -> dinv
    zero_f32  <<<(NNODES + TPB - 1) / TPB, TPB, 0, stream>>>(dinv, NNODES);
    deg_count <<<(NEDGES + TPB - 1) / TPB, TPB, 0, stream>>>(col, dinv, NEDGES);
    make_dinv <<<(NNODES + TPB - 1) / TPB, TPB, 0, stream>>>(dinv, NNODES);

    // 2) pack weights into WMMA B-fragment layout
    pack_W<<<4, TPB, 0, stream>>>(W1, PW1);
    pack_W<<<4, TPB, 0, stream>>>(W2, PW2);

    // ---- Layer 1 ----
    gemm_wmma_bf16<<<NNODES / 16, TPB, 0, stream>>>(x, PW1, XT, NNODES);
    selfloop_bias <<<(NNODES * 32 + TPB - 1) / TPB, TPB, 0, stream>>>(XT, dinv, b1, H, NNODES);
    scatter_edges <<<(NEDGES * 32) / TPB, TPB, 0, stream>>>(row, col, dinv, XT, H, NEDGES);

    // ---- Layer 2 ----  (XT buffer is free again; reuse it)
    gemm_wmma_bf16<<<NNODES / 16, TPB, 0, stream>>>(H, PW2, XT, NNODES);
    selfloop_bias <<<(NNODES * 32 + TPB - 1) / TPB, TPB, 0, stream>>>(XT, dinv, b2, out, NNODES);
    scatter_edges <<<(NEDGES * 32) / TPB, TPB, 0, stream>>>(row, col, dinv, XT, out, NEDGES);
}
